// Block_88201448390974
// MI455X (gfx1250) — matrix operand order repaired and checked by reading, not compiled, not run on hardware
//
#include <hip/hip_runtime.h>
#include <math.h>

// ---------- types ----------
typedef __attribute__((ext_vector_type(16))) __bf16 v16bf;
typedef __attribute__((ext_vector_type(8)))  float  v8f;

#define BN_TOK   16384   // B*N tokens
#define C_DIM    384
#define H_HEADS  6
#define DH       64
#define DFF      1536
#define SEQ      1024
#define BATCH    16

// ---------- WMMA helpers ----------
__device__ __forceinline__ v8f wmma_bf16(v16bf a, v16bf b, v8f c) {
  return __builtin_amdgcn_wmma_f32_16x16x32_bf16(false, a, false, b, (short)0, c,
                                                 false, false);
}

// A-matrix 16x32 bf16 fragment (row-major A, leading dim lda).
// lane<16: row=lane,    K = kc+0..7  and kc+16..23
// lane>=16: row=lane-16, K = kc+8..15 and kc+24..31
__device__ __forceinline__ v16bf load_afrag(const __bf16* A, int lda, int row0,
                                            int kc, int lane) {
  int r  = row0 + (lane & 15);
  int kb = kc + ((lane < 16) ? 0 : 8);
  const __bf16* p = A + (long)r * lda + kb;
  v16bf f;
  ((uint4*)&f)[0] = *(const uint4*)(p);
  ((uint4*)&f)[1] = *(const uint4*)(p + 16);
  return f;
}

// B-matrix 32x16 bf16 fragment where B = W^T, W row-major [N,K]:
// column n of B == row n of W (K = kc+0..7,16..23 for lane<16; kc+8..15,24..31 for lane>=16).
__device__ __forceinline__ v16bf load_bfrag(const __bf16* W, int ldw, int col0,
                                            int kc, int lane) {
  int c  = col0 + (lane & 15);
  int kb = kc + ((lane < 16) ? 0 : 8);
  const __bf16* p = W + (long)c * ldw + kb;
  v16bf f;
  ((uint4*)&f)[0] = *(const uint4*)(p);
  ((uint4*)&f)[1] = *(const uint4*)(p + 16);
  return f;
}

// ---------- CDNA5 async global->LDS copy (ASYNCcnt path) ----------
__device__ __forceinline__ unsigned lds_off(const void* p) {
  return (unsigned)(size_t)p;  // LDS aperture: addr[31:0] is the LDS byte address
}
__device__ __forceinline__ void async_copy_b128(unsigned lds_addr, const void* g) {
  asm volatile("global_load_async_to_lds_b128 %0, %1, off"
               :: "v"(lds_addr), "v"(g) : "memory");
}
__device__ __forceinline__ void wait_async_le1() {
  asm volatile("s_wait_asynccnt 0x1" ::: "memory");
}
__device__ __forceinline__ void wait_async_0() {
  asm volatile("s_wait_asynccnt 0x0" ::: "memory");
}

// ---------- fp32 -> bf16 weight conversion ----------
__global__ __launch_bounds__(256) void cvt_f32_to_bf16(const float* __restrict__ s,
                                                       __bf16* __restrict__ d, int n) {
  int i = blockIdx.x * 256 + threadIdx.x;
  if (i < n) d[i] = (__bf16)s[i];
}

// ---------- LayerNorm (one wave per 384-elem row) -> bf16 ----------
__global__ __launch_bounds__(128) void ln_to_bf16(const float* __restrict__ x,
                                                  const float* __restrict__ g,
                                                  const float* __restrict__ b,
                                                  __bf16* __restrict__ out, int rows) {
  int warp = threadIdx.x >> 5, lane = threadIdx.x & 31;
  int r = blockIdx.x * 4 + warp;
  if (r >= rows) return;
  const float* xr = x + (long)r * C_DIM;
  float v[12];
  float s = 0.f;
#pragma unroll
  for (int j = 0; j < 12; ++j) { v[j] = xr[lane + 32 * j]; s += v[j]; }
#pragma unroll
  for (int m = 16; m >= 1; m >>= 1) s += __shfl_xor(s, m, 32);
  float mean = s * (1.0f / C_DIM);
  float q = 0.f;
#pragma unroll
  for (int j = 0; j < 12; ++j) { float d = v[j] - mean; q += d * d; }
#pragma unroll
  for (int m = 16; m >= 1; m >>= 1) q += __shfl_xor(q, m, 32);
  float rstd = rsqrtf(q * (1.0f / C_DIM) + 1e-5f);
  __bf16* orow = out + (long)r * C_DIM;
#pragma unroll
  for (int j = 0; j < 12; ++j) {
    int c = lane + 32 * j;
    orow[c] = (__bf16)((v[j] - mean) * rstd * g[c] + b[c]);
  }
}

// ---------- NT GEMM: block = 32 rows x 128 cols, 4 waves share an LDS A tile ----
// A tile (32 rows x 32 K, bf16 = 2 KB) is double-buffered in LDS and filled with
// GLOBAL_LOAD_ASYNC_TO_LDS_B128 (one 16B transfer per thread), overlapped with
// the WMMA work on the previous tile via ASYNCcnt.
template <class Epi>
__global__ __launch_bounds__(128) void gemm_nt(const __bf16* __restrict__ A,
                                               const __bf16* __restrict__ W,
                                               int M, int N, int K, Epi epi) {
  __shared__ __bf16 atile[2][32 * 32];
  int warp = threadIdx.x >> 5, lane = threadIdx.x & 31;
  int tilesN = N >> 7;                       // 128-wide block tiles
  int tm = (blockIdx.x / tilesN) << 5;
  int tn = ((blockIdx.x % tilesN) << 7) + (warp << 5);

  // async A-tile loader: thread t moves A[tm + t/4][kc + (t%4)*8 .. +7]
  int t = threadIdx.x;
  int arow = t >> 2, acol = (t & 3) << 3;
  const __bf16* agp = A + (long)(tm + arow) * K + acol;
  unsigned lds0 = lds_off(&atile[0][arow * 32 + acol]);
  unsigned lds1 = lds_off(&atile[1][arow * 32 + acol]);

  async_copy_b128(lds0, agp);                // prefetch kc = 0
  int cur = 0;

  v8f acc00 = {}, acc01 = {}, acc10 = {}, acc11 = {};
  for (int kc = 0; kc < K; kc += 32) {
    bool hasnext = (kc + 32) < K;
    if (hasnext)
      async_copy_b128(cur ? lds0 : lds1, agp + kc + 32);
    if (hasnext) wait_async_le1(); else wait_async_0();
    __syncthreads();

    const __bf16* as = &atile[cur][0];
    v16bf a0 = load_afrag(as, 32, 0, 0, lane);    // rows 0..15 of tile
    v16bf a1 = load_afrag(as, 32, 16, 0, lane);   // rows 16..31 of tile
    v16bf b0 = load_bfrag(W, K, tn, kc, lane);
    v16bf b1 = load_bfrag(W, K, tn + 16, kc, lane);
    acc00 = wmma_bf16(a0, b0, acc00);
    acc01 = wmma_bf16(a0, b1, acc01);
    acc10 = wmma_bf16(a1, b0, acc10);
    acc11 = wmma_bf16(a1, b1, acc11);

    __syncthreads();                         // all reads done before overwrite
    cur ^= 1;
  }

  int nlo = tn + (lane & 15);
  int mb  = (lane >> 4) << 3;  // 0 or 8
#pragma unroll
  for (int i = 0; i < 8; ++i) {
    epi.store(tm + mb + i,      nlo,      acc00[i]);
    epi.store(tm + mb + i,      nlo + 16, acc01[i]);
    epi.store(tm + 16 + mb + i, nlo,      acc10[i]);
    epi.store(tm + 16 + mb + i, nlo + 16, acc11[i]);
  }
}

// ---------- GEMM epilogues ----------
struct EpiQKV {  // scatter to q[B,H,N,Dh] (pre-scaled), k[B,H,N,Dh], vT[B,H,Dh,N]
  __bf16 *q, *k, *vt;
  float scale;
  __device__ void store(int m, int n, float v) const {
    int which = n / C_DIM, c = n % C_DIM;
    int hh = c >> 6, d = c & 63;
    int b = m >> 10, t = m & 1023;
    long idx = ((long)(b * H_HEADS + hh) * SEQ + t) * DH + d;
    if (which == 0)      q[idx] = (__bf16)(v * scale);
    else if (which == 1) k[idx] = (__bf16)v;
    else vt[((long)(b * H_HEADS + hh) * DH + d) * SEQ + t] = (__bf16)v;
  }
};
struct EpiBiasResid {  // out = v + bias[n] + resid[m,n]  (fp32 out)
  const float *bias, *resid;
  float* out;
  __device__ void store(int m, int n, float v) const {
    long i = (long)m * C_DIM + n;
    out[i] = v + bias[n] + resid[i];
  }
};
struct EpiGelu {  // out = gelu(v + bias[n]) * gate[n]  (bf16 out, exact GELU)
  const float *bias, *gate;
  __bf16* out;
  __device__ void store(int m, int n, float v) const {
    float z = v + bias[n];
    float g = 0.5f * z * (1.0f + erff(z * 0.70710678118654752f));
    out[(long)m * DFF + n] = (__bf16)(g * gate[n]);
  }
};

// ---------- flash attention: one wave = 16 query rows of one (b,h) ----------
__global__ __launch_bounds__(128) void attn_kernel(const __bf16* __restrict__ q,
                                                   const __bf16* __restrict__ k,
                                                   const __bf16* __restrict__ vt,
                                                   const float* __restrict__ gate_h,
                                                   __bf16* __restrict__ o) {
  __shared__ __bf16 pbuf[4][16 * 32];
  int warp = threadIdx.x >> 5, lane = threadIdx.x & 31;
  int wid = blockIdx.x * 4 + warp;      // 96*64 waves total
  int bh = wid >> 6;                    // (b,h) pair
  int qt = wid & 63;                    // 16-row query tile
  int b = bh / H_HEADS, h = bh % H_HEADS;
  const __bf16* qb = q  + (long)bh * SEQ * DH;
  const __bf16* kk = k  + (long)bh * SEQ * DH;
  const __bf16* vb = vt + (long)bh * DH * SEQ;

  // Q fragments (q already scaled by Dh^-0.5)
  v16bf qa0 = load_afrag(qb, DH, qt * 16, 0, lane);
  v16bf qa1 = load_afrag(qb, DH, qt * 16, 32, lane);

  v8f acc[4] = {v8f{}, v8f{}, v8f{}, v8f{}};
  float rmax[8], rsum[8];
#pragma unroll
  for (int i = 0; i < 8; ++i) { rmax[i] = -3.0e38f; rsum[i] = 0.f; }

  __bf16* pw = pbuf[warp];
  int col = lane & 15;
  int hi  = lane >> 4;

  for (int kc = 0; kc < SEQ; kc += 32) {
    // S tile 16x32: two 16x16 D-tiles, K-dim(=Dh) split into two WMMAs
    v16bf kb00 = load_bfrag(kk, DH, kc,      0,  lane);
    v16bf kb01 = load_bfrag(kk, DH, kc,      32, lane);
    v16bf kb10 = load_bfrag(kk, DH, kc + 16, 0,  lane);
    v16bf kb11 = load_bfrag(kk, DH, kc + 16, 32, lane);
    v8f z = {};
    v8f s0 = wmma_bf16(qa0, kb00, z);
    s0     = wmma_bf16(qa1, kb01, s0);
    v8f s1 = wmma_bf16(qa0, kb10, z);
    s1     = wmma_bf16(qa1, kb11, s1);

    __syncthreads();  // pbuf reuse fence (all waves same trip count)
#pragma unroll
    for (int i = 0; i < 8; ++i) {
      float a = s0[i], c = s1[i];
      float mx = fmaxf(a, c);
#pragma unroll
      for (int m = 8; m >= 1; m >>= 1) mx = fmaxf(mx, __shfl_xor(mx, m, 32));
      float mn = fmaxf(rmax[i], mx);
      float alpha = __expf(rmax[i] - mn);
      rmax[i] = mn;
      float p0 = __expf(a - mn), p1 = __expf(c - mn);
      float rs = p0 + p1;
#pragma unroll
      for (int m = 8; m >= 1; m >>= 1) rs += __shfl_xor(rs, m, 32);
      rsum[i] = rsum[i] * alpha + rs;
#pragma unroll
      for (int t = 0; t < 4; ++t) acc[t][i] *= alpha;
      int rl = (hi << 3) + i;
      pw[rl * 32 + col]      = (__bf16)p0;
      pw[rl * 32 + col + 16] = (__bf16)p1;
    }
    __syncthreads();

    // Re-read P as A-fragment (16x32) from LDS
    v16bf pf;
    {
      const __bf16* pp = pw + col * 32 + (hi ? 8 : 0);
      ((uint4*)&pf)[0] = *(const uint4*)(pp);
      ((uint4*)&pf)[1] = *(const uint4*)(pp + 16);
    }
    // PV: B = V (32 keys x 64 d) via V^T rows (contiguous)
#pragma unroll
    for (int t = 0; t < 4; ++t) {
      const __bf16* vp = vb + (long)(t * 16 + col) * SEQ + kc + (hi ? 8 : 0);
      v16bf bf;
      ((uint4*)&bf)[0] = *(const uint4*)(vp);
      ((uint4*)&bf)[1] = *(const uint4*)(vp + 16);
      acc[t] = wmma_bf16(pf, bf, acc[t]);
    }
  }

  float gh = gate_h[h];
#pragma unroll
  for (int i = 0; i < 8; ++i) {
    float inv = gh / rsum[i];
    int row = qt * 16 + (hi << 3) + i;
    long base = ((long)b * SEQ + row) * C_DIM + h * DH;
#pragma unroll
    for (int t = 0; t < 4; ++t)
      o[base + t * 16 + col] = (__bf16)(acc[t][i] * inv);
  }
}

// ---------- host ----------
static inline size_t alignup(size_t x) { return (x + 255) & ~(size_t)255; }

extern "C" void kernel_launch(void* const* d_in, const int* in_sizes, int n_in,
                              void* d_out, int out_size, void* d_ws, size_t ws_size,
                              hipStream_t stream) {
  const float* x       = (const float*)d_in[0];
  const float* ln1_g   = (const float*)d_in[1];
  const float* ln1_b   = (const float*)d_in[2];
  const float* qkv_w   = (const float*)d_in[3];
  const float* proj_w  = (const float*)d_in[4];
  const float* proj_b  = (const float*)d_in[5];
  const float* gate_h  = (const float*)d_in[6];
  const float* ln2_g   = (const float*)d_in[7];
  const float* ln2_b   = (const float*)d_in[8];
  const float* fc1_w   = (const float*)d_in[9];
  const float* fc1_b   = (const float*)d_in[10];
  const float* fc2_w   = (const float*)d_in[11];
  const float* fc2_b   = (const float*)d_in[12];
  const float* gate_mlp= (const float*)d_in[13];
  float* out = (float*)d_out;

  // workspace carve
  char* p = (char*)d_ws;
  auto take = [&](size_t bytes) { char* r = p; p += alignup(bytes); return r; };
  __bf16* wqkv = (__bf16*)take((size_t)3 * C_DIM * C_DIM * 2);
  __bf16* wproj= (__bf16*)take((size_t)C_DIM * C_DIM * 2);
  __bf16* wfc1 = (__bf16*)take((size_t)DFF * C_DIM * 2);
  __bf16* wfc2 = (__bf16*)take((size_t)C_DIM * DFF * 2);
  __bf16* hbf  = (__bf16*)take((size_t)BN_TOK * C_DIM * 2);
  __bf16* qbuf = (__bf16*)take((size_t)BN_TOK * C_DIM * 2);
  __bf16* kbuf = (__bf16*)take((size_t)BN_TOK * C_DIM * 2);
  __bf16* vtbuf= (__bf16*)take((size_t)BN_TOK * C_DIM * 2);
  __bf16* obuf = (__bf16*)take((size_t)BN_TOK * C_DIM * 2);
  float*  x1   = (float*) take((size_t)BN_TOK * C_DIM * 4);
  __bf16* h2   = (__bf16*)take((size_t)BN_TOK * C_DIM * 2);
  __bf16* mbuf = (__bf16*)take((size_t)BN_TOK * DFF * 2);

  // weight conversion fp32 -> bf16
  auto cvt = [&](const float* s, __bf16* d, int n) {
    cvt_f32_to_bf16<<<(n + 255) / 256, 256, 0, stream>>>(s, d, n);
  };
  cvt(qkv_w, wqkv, 3 * C_DIM * C_DIM);
  cvt(proj_w, wproj, C_DIM * C_DIM);
  cvt(fc1_w, wfc1, DFF * C_DIM);
  cvt(fc2_w, wfc2, C_DIM * DFF);

  // LN1 -> bf16
  ln_to_bf16<<<BN_TOK / 4, 128, 0, stream>>>(x, ln1_g, ln1_b, hbf, BN_TOK);

  // QKV GEMM: [16384,384] x [1152,384]^T, scatter epilogue
  {
    EpiQKV epi{qbuf, kbuf, vtbuf, 0.125f};  // Dh^-0.5 = 1/8
    int blocks = (BN_TOK / 32) * ((3 * C_DIM) / 128);
    gemm_nt<EpiQKV><<<blocks, 128, 0, stream>>>(hbf, wqkv, BN_TOK, 3 * C_DIM,
                                                C_DIM, epi);
  }

  // attention: 96 (b,h) * 64 query tiles = 6144 waves
  attn_kernel<<<(BATCH * H_HEADS * 64) / 4, 128, 0, stream>>>(qbuf, kbuf, vtbuf,
                                                              gate_h, obuf);

  // proj GEMM + bias + residual -> x1 (fp32)
  {
    EpiBiasResid epi{proj_b, x, x1};
    int blocks = (BN_TOK / 32) * (C_DIM / 128);
    gemm_nt<EpiBiasResid><<<blocks, 128, 0, stream>>>(obuf, wproj, BN_TOK,
                                                      C_DIM, C_DIM, epi);
  }

  // LN2 -> bf16
  ln_to_bf16<<<BN_TOK / 4, 128, 0, stream>>>(x1, ln2_g, ln2_b, h2, BN_TOK);

  // FC1 GEMM + bias + GELU + gate -> bf16
  {
    EpiGelu epi{fc1_b, gate_mlp, mbuf};
    int blocks = (BN_TOK / 32) * (DFF / 128);
    gemm_nt<EpiGelu><<<blocks, 128, 0, stream>>>(h2, wfc1, BN_TOK, DFF, C_DIM,
                                                 epi);
  }

  // FC2 GEMM + bias + residual -> out (fp32)
  {
    EpiBiasResid epi{fc2_b, x1, out};
    int blocks = (BN_TOK / 32) * (C_DIM / 128);
    gemm_nt<EpiBiasResid><<<blocks, 128, 0, stream>>>(mbuf, wfc2, BN_TOK,
                                                      C_DIM, DFF, epi);
  }
}
